// GATConv_30743375904932
// MI455X (gfx1250) — matrix operand order repaired and checked by reading, not compiled, not run on hardware
//
#include <hip/hip_runtime.h>
#include <hip/hip_bf16.h>
#include <cstdint>
#include <cstddef>

// ---------------------------------------------------------------------------
// GAT layer, fused for MI455X (gfx1250, wave32, WMMA).
//   N = 8192 nodes, F = 128 features.
//   Dominant cost: streaming dense A (256 MB int32) exactly once (~11 us at
//   23.3 TB/s). alpha @ H (17.2 GFLOP) runs on v_wmma_f32_16x16x32_f16.
// ---------------------------------------------------------------------------

typedef _Float16 v16h __attribute__((ext_vector_type(16)));
typedef float    v8f  __attribute__((ext_vector_type(8)));

#define N_NODES 8192
#define F_DIM   128
#define LEAKY   0.2f

union FragU { v16h h; unsigned u[8]; };

// ---------------------------------------------------------------------------
// Kernel 1: H = X @ W (f32 in registers), write H^T as f16 [F][N],
//           es = H @ a_src, ed = H @ a_dst.
// One block = 4 node rows, 128 threads (one per output feature).
// ---------------------------------------------------------------------------
__global__ __launch_bounds__(128) void gat_proj_kernel(
    const float* __restrict__ X, const float* __restrict__ W,
    const float* __restrict__ a_src, const float* __restrict__ a_dst,
    _Float16* __restrict__ HhT, float* __restrict__ es, float* __restrict__ ed)
{
  __shared__ float sX[4 * 128];
  __shared__ float red[128];
  const int f  = threadIdx.x;
  const int i0 = blockIdx.x * 4;

  #pragma unroll
  for (int r = 0; r < 4; ++r)
    sX[r * 128 + f] = X[(size_t)(i0 + r) * F_DIM + f];
  __syncthreads();

  float h0 = 0.f, h1 = 0.f, h2 = 0.f, h3 = 0.f;
  for (int k = 0; k < F_DIM; ++k) {
    const float wv = W[k * F_DIM + f];           // coalesced across f
    h0 = fmaf(sX[k],       wv, h0);
    h1 = fmaf(sX[128 + k], wv, h1);
    h2 = fmaf(sX[256 + k], wv, h2);
    h3 = fmaf(sX[384 + k], wv, h3);
  }

  // H^T[f][i0..i0+3] as f16 (8-byte coalesced store)
  union { uint2 u; _Float16 h[4]; } pk;
  pk.h[0] = (_Float16)h0; pk.h[1] = (_Float16)h1;
  pk.h[2] = (_Float16)h2; pk.h[3] = (_Float16)h3;
  *(uint2*)(HhT + (size_t)f * N_NODES + i0) = pk.u;

  const float hs[4] = {h0, h1, h2, h3};
  const float asf = a_src[f], adf = a_dst[f];
  #pragma unroll
  for (int r = 0; r < 4; ++r) {
    red[f] = hs[r] * asf; __syncthreads();
    for (int st = 64; st > 0; st >>= 1) { if (f < st) red[f] += red[f + st]; __syncthreads(); }
    if (f == 0) es[i0 + r] = red[0];
    __syncthreads();
    red[f] = hs[r] * adf; __syncthreads();
    for (int st = 64; st > 0; st >>= 1) { if (f < st) red[f] += red[f + st]; __syncthreads(); }
    if (f == 0) ed[i0 + r] = red[0];
    __syncthreads();
  }
}

// ---------------------------------------------------------------------------
// Kernel 2: M = max_j ed[j]  (global softmax shift bound; leaky is monotone)
// ---------------------------------------------------------------------------
__global__ __launch_bounds__(256) void gat_maxed_kernel(
    const float* __restrict__ ed, float* __restrict__ Mout)
{
  __shared__ float r[256];
  float m = -3.0e38f;
  for (int i = threadIdx.x; i < N_NODES; i += 256) m = fmaxf(m, ed[i]);
  r[threadIdx.x] = m; __syncthreads();
  for (int st = 128; st > 0; st >>= 1) {
    if (threadIdx.x < st) r[threadIdx.x] = fmaxf(r[threadIdx.x], r[threadIdx.x + st]);
    __syncthreads();
  }
  if (threadIdx.x == 0) Mout[0] = r[0];
}

// ---------------------------------------------------------------------------
// Kernel 3: fused masked softmax + aggregation + ELU.
// Block = 256 threads = 8 waves, covers 32 output rows (two 16-row WMMA
// tiles). Waves 0-3 -> row-tile 0, waves 4-7 -> row-tile 1; each wave owns
// 32 features (two 16x16x32 WMMAs per K-step). K-loop streams all 8192
// neighbors in tiles of 32: A read once, p = exp(sel(mask, e-s, -1024))
// (branchless, masked -> exact 0) built in LDS, H tile staged feature-major
// in LDS (stride 40 halves, bank-friendly).
// ---------------------------------------------------------------------------
__global__ __launch_bounds__(256) void gat_attn_kernel(
    const int* __restrict__ A, const _Float16* __restrict__ HhT,
    const float* __restrict__ es, const float* __restrict__ ed,
    const float* __restrict__ Mp, float* __restrict__ out)
{
  __shared__ alignas(16) _Float16 HT[128 * 40];     // H tile, [feature][k], padded
  __shared__ alignas(16) _Float16 P[2 * 16 * 40];   // two 16x32 p tiles, padded
  __shared__ float zbuf[256];
  __shared__ float Zinv[32];
  __shared__ float srow[32];

  const int t    = threadIdx.x;
  const int wave = t >> 5;
  const int lane = t & 31;
  const int hi   = lane >> 4;     // lane half (WMMA layout)
  const int lm   = lane & 15;

  // p-compute role: one row, 4 adjacency columns per thread
  const int row32 = t >> 3;       // 0..31
  const int jq    = t & 7;        // 4-col group
  const int pg    = row32 >> 4;   // which 16-row tile
  const int ii    = row32 & 15;

  // H staging role
  const int f    = t >> 1;        // feature 0..127
  const int part = t & 1;

  // WMMA role
  const int wg = wave >> 2;       // row-tile this wave accumulates
  const int wl = wave & 3;        // 32-feature slice within tile

  const int ibase = blockIdx.x * 32;
  const float Mval = Mp[0];
  if (t < 32) {
    const float e = es[ibase + t] + Mval;
    srow[t] = e > 0.f ? e : LEAKY * e;              // upper bound on row logits
  }
  __syncthreads();

  const float esv  = es[ibase + row32];
  const float s    = srow[row32];
  const int   rowA = ibase + row32;
  const int4*  Arow = (const int4*)(A + (size_t)rowA * N_NODES);
  const uint4* Hrow = (const uint4*)(HhT + (size_t)f * N_NODES);

  float z = 0.f;
  v8f acc0, acc1;
  #pragma unroll
  for (int r = 0; r < 8; ++r) { acc0[r] = 0.f; acc1[r] = 0.f; }

  for (int jt = 0; jt < N_NODES / 32; ++jt) {
    const int jbase = jt * 32;

    // --- stage 32x128 H tile (feature-major) into LDS: 16 halves/thread ---
    const uint4 q0 = Hrow[(jbase >> 3) + part * 2];
    const uint4 q1 = Hrow[(jbase >> 3) + part * 2 + 1];
    *(uint4*)&HT[f * 40 + part * 16]     = q0;
    *(uint4*)&HT[f * 40 + part * 16 + 8] = q1;

    // --- attention weights for 2 x (16x32) tile: 4 columns/thread ---
    const int  j0 = jbase + 4 * jq;
    const int4 av = Arow[(jbase >> 2) + jq];        // coalesced 16B loads of A
    // keep the HBM stream of A rolling (~1KB / 8 tiles ahead)
    {
      const int pj = (jbase + 256 <= N_NODES - 32) ? (jbase + 256) : (N_NODES - 32);
      __builtin_prefetch(&Arow[(pj >> 2) + jq], 0, 3);
    }
    const float4 edv = *(const float4*)(ed + j0);   // one b128, L2-resident
    const float  pe[4] = {edv.x, edv.y, edv.z, edv.w};
    const int    am[4] = {av.x, av.y, av.z, av.w};
    float pv[4];
    #pragma unroll
    for (int c = 0; c < 4; ++c) {
      const float er  = esv + pe[c];
      const float e   = er > 0.f ? er : LEAKY * er;        // LeakyReLU(0.2)
      const bool  mk  = (am[c] > 0) | ((j0 + c) == rowA);  // A>0 or self-loop
      const float arg = mk ? (e - s) : -1024.f;            // branchless mask
      const float p   = __expf(arg);                       // masked -> exact 0
      pv[c] = p;
      z += p;
    }
    union { uint2 u; _Float16 h[4]; } pq;
    #pragma unroll
    for (int c = 0; c < 4; ++c) pq.h[c] = (_Float16)pv[c];
    *(uint2*)&P[pg * 640 + ii * 40 + 4 * jq] = pq.u;
    __syncthreads();

    // --- WMMA: acc += P_tile(16x32) @ H_tile(32x16|16) ---
    FragU af, b0, b1;
    const unsigned* Pw = (const unsigned*)(P + wg * 640);
    const int arb = lm * 20 + hi * 8;               // A 16x32 f16 layout
    #pragma unroll
    for (int v = 0; v < 4; ++v) af.u[v]     = Pw[arb + v];
    #pragma unroll
    for (int v = 0; v < 4; ++v) af.u[4 + v] = Pw[arb + 4 + v];

    const unsigned* Hw = (const unsigned*)HT;
    const int brb0 = (wl * 32 + lm) * 20 + hi * 8;  // B 32x16 f16 layout
    const int brb1 = brb0 + 16 * 20;
    #pragma unroll
    for (int v = 0; v < 8; ++v) b0.u[v] = Hw[brb0 + v];
    #pragma unroll
    for (int v = 0; v < 8; ++v) b1.u[v] = Hw[brb1 + v];

    acc0 = __builtin_amdgcn_wmma_f32_16x16x32_f16(false, af.h, false, b0.h,
                                                  (short)0, acc0, false, false);
    acc1 = __builtin_amdgcn_wmma_f32_16x16x32_f16(false, af.h, false, b1.h,
                                                  (short)0, acc1, false, false);
    __syncthreads();
  }

  // --- reduce softmax denominators (8 partials per row), invert once ---
  zbuf[t] = z; __syncthreads();
  if (t < 32) {
    float zz = 0.f;
    #pragma unroll
    for (int q = 0; q < 8; ++q) zz += zbuf[t * 8 + q];
    Zinv[t] = 1.f / zz;                             // > 0: self-loop guaranteed
  }
  __syncthreads();

  // --- normalize, ELU, store (C layout: VGPR r -> M = hi*8 + r, N = lm) ---
  const int n0 = wl * 32 + lm;
  #pragma unroll
  for (int r = 0; r < 8; ++r) {
    const int mr  = wg * 16 + hi * 8 + r;
    const size_t ro = (size_t)(ibase + mr) * F_DIM;
    const float rz = Zinv[mr];
    float v0 = acc0[r] * rz;
    float v1 = acc1[r] * rz;
    v0 = v0 > 0.f ? v0 : expm1f(v0);
    v1 = v1 > 0.f ? v1 : expm1f(v1);
    out[ro + n0]      = v0;
    out[ro + n0 + 16] = v1;
  }
}

// ---------------------------------------------------------------------------
extern "C" void kernel_launch(void* const* d_in, const int* in_sizes, int n_in,
                              void* d_out, int out_size, void* d_ws, size_t ws_size,
                              hipStream_t stream) {
  (void)in_sizes; (void)n_in; (void)out_size; (void)ws_size;
  const float* X     = (const float*)d_in[0];
  const int*   A     = (const int*)d_in[1];
  const float* W     = (const float*)d_in[2];
  const float* a_src = (const float*)d_in[3];
  const float* a_dst = (const float*)d_in[4];
  float* out = (float*)d_out;

  char* ws = (char*)d_ws;
  _Float16* HhT = (_Float16*)ws;                              // 2 MB, 16B aligned
  float* es = (float*)(ws + (size_t)F_DIM * N_NODES * 2);     // 32 KB
  float* ed = es + N_NODES;                                   // 32 KB
  float* Mv = ed + N_NODES;                                   // 4 B

  gat_proj_kernel <<<N_NODES / 4, 128, 0, stream>>>(X, W, a_src, a_dst, HhT, es, ed);
  gat_maxed_kernel<<<1,           256, 0, stream>>>(ed, Mv);
  gat_attn_kernel <<<N_NODES / 32, 256, 0, stream>>>(A, HhT, es, ed, Mv, out);
}
